// TorchModel_21741124452502
// MI455X (gfx1250) — matrix operand order repaired and checked by reading, not compiled, not run on hardware
//
#include <hip/hip_runtime.h>
#include <hip/hip_bf16.h>

// ---------------------------------------------------------------------------
// Problem constants (from reference): B=32768, IN=64, H=256, OUT=768
// BM=32 batch rows per workgroup: each wave computes two 16-row M tiles that
// share every B (weight) fragment -> halves L2 weight traffic vs BM=16.
// ---------------------------------------------------------------------------
#define B_SZ 32768
#define BM   32
#define LDA  520          // LDS row stride (bf16 elems) for activation tiles

typedef __attribute__((ext_vector_type(16))) __bf16 v16bf;
typedef __attribute__((ext_vector_type(8)))  float  v8f;

// Packed-weight geometry: frags are 32(K) x 16(N), 512 bf16, lane-major
#define W1_FRAGS (10 * 64)            // K=320, N=1024 (4 gates concat)
#define W2_FRAGS (16 * 64)            // K=512, N=1024
#define FF_FRAGS (8 * 48)             // K=256, N=768
#define W1_ELEMS (W1_FRAGS * 512)     // 327680
#define W2_ELEMS (W2_FRAGS * 512)     // 524288
#define FF_ELEMS (FF_FRAGS * 512)     // 196608
#define PACK_ELEMS (W1_ELEMS + W2_ELEMS + FF_ELEMS)   // 1048576

// d_out layout (flat, return order): means, deviances, mult, new_state1, new_state2
#define OFF_DEV  ((size_t)B_SZ * 256)
#define OFF_MULT ((size_t)B_SZ * 512)
#define OFF_S1   ((size_t)B_SZ * 768)
#define OFF_S2   ((size_t)B_SZ * 768 + (size_t)B_SZ * 512)

struct WPtrs {
  const float* l1w[4]; const float* l1b[4];
  const float* l2w[4]; const float* l2b[4];
  const float* ffw;
};

__device__ __forceinline__ unsigned short f2bf(float f) {
  unsigned int u = __float_as_uint(f);
  u += 0x7FFFu + ((u >> 16) & 1u);     // round-to-nearest-even
  return (unsigned short)(u >> 16);
}

__device__ __forceinline__ float sigmoidf_(float x) {
  return 1.0f / (1.0f + __expf(-x));
}

// ---------------------------------------------------------------------------
// Weight pre-pack: fp32 row-major [K,N-per-gate] -> bf16 WMMA B-fragment order.
// Fragment (kt,nt): lane l, elem s (=2v+j) holds W[kt*32 + (l>>4)*16 + s,
//                                               nt*16 + (l&15)]
// ---------------------------------------------------------------------------
__global__ __launch_bounds__(256) void pack_weights(WPtrs p,
                                                    unsigned short* __restrict__ wp,
                                                    float* __restrict__ b1c,
                                                    float* __restrict__ b2c) {
  const int i = blockIdx.x * 256 + threadIdx.x;   // exactly PACK_ELEMS threads
  float v;
  if (i < W1_ELEMS) {
    int e = i, frag = e >> 9, w = e & 511;
    int lane = w >> 4, sub = w & 15;
    int kt = frag >> 6, nt = frag & 63;
    int k = kt * 32 + (lane >> 4) * 16 + sub;
    int n = nt * 16 + (lane & 15);
    v = p.l1w[n >> 8][k * 256 + (n & 255)];
  } else if (i < W1_ELEMS + W2_ELEMS) {
    int e = i - W1_ELEMS, frag = e >> 9, w = e & 511;
    int lane = w >> 4, sub = w & 15;
    int kt = frag >> 6, nt = frag & 63;
    int k = kt * 32 + (lane >> 4) * 16 + sub;
    int n = nt * 16 + (lane & 15);
    v = p.l2w[n >> 8][k * 256 + (n & 255)];
  } else {
    int e = i - W1_ELEMS - W2_ELEMS, frag = e >> 9, w = e & 511;
    int lane = w >> 4, sub = w & 15;
    int kt = frag / 48, nt = frag % 48;
    int k = kt * 32 + (lane >> 4) * 16 + sub;
    int n = nt * 16 + (lane & 15);
    v = p.ffw[k * 768 + n];
  }
  wp[i] = f2bf(v);

  if (blockIdx.x == 0) {                 // combined biases (fp32, tiny)
    for (int j = threadIdx.x; j < 1024; j += 256) {
      b1c[j] = p.l1b[j >> 8][j & 255];
      b2c[j] = p.l2b[j >> 8][j & 255];
    }
  }
}

// ---------------------------------------------------------------------------
// Wave-level GEMM core, 2 M-tiles (rows 0-15 and 16-31 of the block tile)
// sharing every B fragment. A from LDS (bf16, stride LDA); B from packed
// global (L2-resident); C in v8f tiles.
// ---------------------------------------------------------------------------
template <int KT, int NTW>
__device__ __forceinline__ void gemm_bf16x2(const unsigned short* __restrict__ wp,
                                            const unsigned short* xb,
                                            int lane, int nt0, int NT,
                                            v8f* acc0, v8f* acc1) {
  const int row = lane & 15;
  const int kg  = lane >> 4;
  for (int kt = 0; kt < KT; ++kt) {
    union { uint4 u[2]; v16bf v; } fa0, fa1;
    // A fragment: lanes 0-15 get K 0..15, lanes 16-31 get K 16..31
    fa0.u[0] = *(const uint4*)(xb + row * LDA + kt * 32 + kg * 16);
    fa0.u[1] = *(const uint4*)(xb + row * LDA + kt * 32 + 8 + kg * 16);
    fa1.u[0] = *(const uint4*)(xb + (row + 16) * LDA + kt * 32 + kg * 16);
    fa1.u[1] = *(const uint4*)(xb + (row + 16) * LDA + kt * 32 + 8 + kg * 16);
    if (kt + 1 < KT)  // warm GL2 for next K-step's weight panel
      __builtin_prefetch(wp + ((size_t)((kt + 1) * NT + nt0) << 9) + lane * 16, 0, 1);
#pragma unroll
    for (int j = 0; j < NTW; ++j) {
      const unsigned short* bp =
          wp + ((size_t)(kt * NT + nt0 + j) << 9) + lane * 16;
      union { uint4 u[2]; v16bf v; } fb;
      fb.u[0] = *(const uint4*)bp;
      fb.u[1] = *(const uint4*)(bp + 8);
      acc0[j] = __builtin_amdgcn_wmma_f32_16x16x32_bf16(
          false, fa0.v, false, fb.v, (short)0, acc0[j], false, false);
      acc1[j] = __builtin_amdgcn_wmma_f32_16x16x32_bf16(
          false, fa1.v, false, fb.v, (short)0, acc1[j], false, false);
    }
  }
}

// ---------------------------------------------------------------------------
// Fused 2-layer LSTM step + FF head. 32 batch rows / block, 8 wave32s.
// Gate epilogues run in two 16-row halves to keep the f32 LDS stage at 64KB.
// ---------------------------------------------------------------------------
__global__ __launch_bounds__(256) void lstm_fused(
    const float* __restrict__ inp, const float* __restrict__ s1,
    const float* __restrict__ s2, const unsigned short* __restrict__ wpack,
    const float* __restrict__ b1c, const float* __restrict__ b2c,
    float* __restrict__ out) {
  __shared__ unsigned short xbuf[BM * LDA];   // bf16 activation tile (K<=512)
  __shared__ float gact[16 * 1024];           // f32 gate/FF staging (one half)

  const int tid  = threadIdx.x;
  const int lane = tid & 31;
  const int wave = tid >> 5;
  const int r0   = blockIdx.x * BM;
  const int kg   = lane >> 4;

  // ---- stage L1 input: x1 = [inp | state1[:, :256]] as bf16 ----
  for (int i = tid; i < BM * 320; i += 256) {
    int r = i / 320, c = i % 320;
    float v = (c < 64) ? inp[(size_t)(r0 + r) * 64 + c]
                       : s1[(size_t)(r0 + r) * 512 + (c - 64)];
    xbuf[r * LDA + c] = f2bf(v);
  }
  __syncthreads();

  v8f acc0[8], acc1[8];
  const v8f vz = {0.f, 0.f, 0.f, 0.f, 0.f, 0.f, 0.f, 0.f};

  // =============== LSTM layers (layer = 0, 1) ===============
#pragma unroll 1
  for (int layer = 0; layer < 2; ++layer) {
    const float* bc    = layer ? b2c : b1c;
    const float* st    = layer ? s2 : s1;
    const size_t offst = layer ? OFF_S2 : OFF_S1;

#pragma unroll
    for (int j = 0; j < 8; ++j) { acc0[j] = vz; acc1[j] = vz; }
    if (layer == 0)
      gemm_bf16x2<10, 8>(wpack, xbuf, lane, wave * 8, 64, acc0, acc1);
    else
      gemm_bf16x2<16, 8>(wpack + W1_ELEMS, xbuf, lane, wave * 8, 64, acc0, acc1);

#pragma unroll
    for (int h = 0; h < 2; ++h) {
      const v8f* acc = h ? acc1 : acc0;
      // activated gates of this 16-row half -> gact
#pragma unroll
      for (int j = 0; j < 8; ++j) {
        int n = wave * 128 + j * 16 + (lane & 15);
        int g = n >> 8;                       // 0=f 1=k 2=i(tanh) 3=s
#pragma unroll
        for (int v = 0; v < 8; ++v) {
          int m = kg * 8 + v;
          float x = acc[j][v] + bc[n];
          gact[m * 1024 + n] = (g == 2) ? tanhf(x) : sigmoidf_(x);
        }
      }
      __syncthreads();
      // combine gates -> cell/out, emit new_state, restage next layer's A
      for (int i = tid; i < 16 * 256; i += 256) {
        int r = i >> 8, n = i & 255;
        int gr = h * 16 + r;                  // row within 32-row block tile
        float reset  = gact[r * 1024 + n];
        float wrt    = gact[r * 1024 + 256 + n];
        float interm = gact[r * 1024 + 512 + n];
        float readg  = gact[r * 1024 + 768 + n];
        float cprev  = st[(size_t)(r0 + gr) * 512 + 256 + n];
        float cell   = reset * cprev + wrt * interm;
        float o      = readg * tanhf(cell);
        out[offst + (size_t)(r0 + gr) * 512 + n]       = o;
        out[offst + (size_t)(r0 + gr) * 512 + 256 + n] = cell;
        xbuf[gr * LDA + n] = f2bf(o);         // next GEMM's A (cols 0..255)
        if (layer == 0)                        // L2 A: append state2 prev_out
          xbuf[gr * LDA + 256 + n] = f2bf(s2[(size_t)(r0 + gr) * 512 + n]);
      }
      __syncthreads();
    }
  }

  // ======================= FF head =======================
#pragma unroll
  for (int j = 0; j < 6; ++j) { acc0[j] = vz; acc1[j] = vz; }
  gemm_bf16x2<8, 6>(wpack + W1_ELEMS + W2_ELEMS, xbuf, lane, wave * 6, 48,
                    acc0, acc1);

#pragma unroll
  for (int h = 0; h < 2; ++h) {
    const v8f* acc = h ? acc1 : acc0;
#pragma unroll
    for (int j = 0; j < 6; ++j) {
      int n = wave * 96 + j * 16 + (lane & 15);
#pragma unroll
      for (int v = 0; v < 8; ++v) {
        int m = kg * 8 + v;
        gact[m * 1024 + n] = acc[j][v];
      }
    }
    __syncthreads();

    // means + deviances for this half
    for (int i = tid; i < 16 * 256; i += 256) {
      int r = i >> 8, n = i & 255;
      int gr = h * 16 + r;
      out[(size_t)(r0 + gr) * 256 + n]           = gact[r * 1024 + n];
      out[OFF_DEV + (size_t)(r0 + gr) * 256 + n] = __expf(gact[r * 1024 + 256 + n]);
    }
    // mult: softmax over HM=4 heads at each of TS=64 slots
    for (int i = tid; i < 16 * 64; i += 256) {
      int r = i >> 6, t = i & 63;
      int gr = h * 16 + r;
      float a0 = gact[r * 1024 + 512 + t];
      float a1 = gact[r * 1024 + 576 + t];
      float a2 = gact[r * 1024 + 640 + t];
      float a3 = gact[r * 1024 + 704 + t];
      float mx = fmaxf(fmaxf(a0, a1), fmaxf(a2, a3));
      float e0 = __expf(a0 - mx), e1 = __expf(a1 - mx);
      float e2 = __expf(a2 - mx), e3 = __expf(a3 - mx);
      float rs = 1.0f / (e0 + e1 + e2 + e3);
      size_t base = OFF_MULT + (size_t)(r0 + gr) * 256 + t;
      out[base]       = e0 * rs;
      out[base + 64]  = e1 * rs;
      out[base + 128] = e2 * rs;
      out[base + 192] = e3 * rs;
    }
    __syncthreads();
  }
}

// ---------------------------------------------------------------------------
extern "C" void kernel_launch(void* const* d_in, const int* in_sizes, int n_in,
                              void* d_out, int out_size, void* d_ws, size_t ws_size,
                              hipStream_t stream) {
  (void)in_sizes; (void)n_in; (void)out_size; (void)ws_size;
  // setup_inputs() dict order: inp, state1, state2, then per gate g in {f,k,i,s}:
  //   l1_wg, l1_bg, l2_wg, l2_bg  (interleaved!), then ff_w.
  WPtrs p;
  for (int g = 0; g < 4; ++g) {
    p.l1w[g] = (const float*)d_in[3 + 4 * g];
    p.l1b[g] = (const float*)d_in[4 + 4 * g];
    p.l2w[g] = (const float*)d_in[5 + 4 * g];
    p.l2b[g] = (const float*)d_in[6 + 4 * g];
  }
  p.ffw = (const float*)d_in[19];

  unsigned short* wpack = (unsigned short*)d_ws;                    // 2 MB bf16
  float* b1c = (float*)((char*)d_ws + (size_t)PACK_ELEMS * 2);      // 4 KB
  float* b2c = b1c + 1024;                                          // 4 KB

  pack_weights<<<PACK_ELEMS / 256, 256, 0, stream>>>(p, wpack, b1c, b2c);
  lstm_fused<<<B_SZ / BM, 256, 0, stream>>>(
      (const float*)d_in[0], (const float*)d_in[1], (const float*)d_in[2],
      wpack, b1c, b2c, (float*)d_out);
}
